// VC_Aggregator_85048942395941
// MI455X (gfx1250) — matrix operand order repaired and checked by reading, not compiled, not run on hardware
//
#include <hip/hip_runtime.h>
#include <hip/hip_bf16.h>

typedef __attribute__((ext_vector_type(16))) _Float16 v16h;
typedef __attribute__((ext_vector_type(4)))  _Float16 v4h;
typedef __attribute__((ext_vector_type(8)))  float    v8f;
typedef __attribute__((ext_vector_type(4)))  float    v4f;

#define BDIM 4096
#define LDIM 50
#define DDIM 128
#define MROWS 64          // L padded to 64 -> 4 WMMA M-tiles
#define X0_STRIDE 264     // 256 + 8 halves padding (33*16B rows -> b128-aligned)
#define X1_STRIDE 136     // 128 + 8 halves padding (17*16B rows)

// f16 weight layout inside d_ws (element offsets, halves):
#define W1H_OFF 0                       // 128*256
#define W2H_OFF (W1H_OFF + 128 * 256)   // 128*128
#define A1H_OFF (W2H_OFF + 128 * 128)   // 128*256
#define A2H_OFF (A1H_OFF + 128 * 256)   // 128*128
#define WSH_TOTAL (A2H_OFF + 128 * 128) // 98304 halves = 192KB

// ---- One-shot weight conversion f32 -> f16 into workspace (vectorized x4) ----
__global__ __launch_bounds__(256)
void cvt_weights_kernel(const float* __restrict__ src, _Float16* __restrict__ dst,
                        int n4) {
  int i = blockIdx.x * blockDim.x + threadIdx.x;
  if (i < n4) {
    v4f v = *(const v4f*)(src + i * 4);
    v4h h;
    h[0] = (_Float16)v[0]; h[1] = (_Float16)v[1];
    h[2] = (_Float16)v[2]; h[3] = (_Float16)v[3];
    *(v4h*)(dst + i * 4) = h;
  }
}

// ---- A fragment: 16x32 f16, lanes 0-15 rows M=0..15 (lanes 16-31 same rows,
// K offset +8/+24). Two contiguous 8-half (16B) LDS reads per lane. ----
__device__ __forceinline__ v16h load_a_frag(const _Float16* X, int stride,
                                            int row, int kbase, int lane) {
  const _Float16* p = X + row * stride + kbase + ((lane >> 4) << 3);
  v16h a;
#pragma unroll
  for (int e = 0; e < 8; ++e)  a[e] = p[e];
#pragma unroll
  for (int e = 8; e < 16; ++e) a[e] = p[e + 8];   // K = base+16 .. base+23
  return a;
}

// ---- B fragment: 32x16 f16, col = lane%16 = output feature d,
// K = (lane/16)*8 + e (e<8), 16 + (lane/16)*8 + (e-8) (e>=8) of W_h[d][*]. ----
__device__ __forceinline__ v16h load_b_frag(const _Float16* __restrict__ W, int Kdim,
                                            int d, int kc, int lane) {
  const _Float16* p = W + (size_t)d * Kdim + kc * 32 + ((lane >> 4) << 3);
  v16h b;
#pragma unroll
  for (int e = 0; e < 8; ++e)  b[e] = p[e];
#pragma unroll
  for (int e = 8; e < 16; ++e) b[e] = p[e + 8];
  return b;
}

// ---- Dense layer: Xout[64, D] = relu(W @ Xin + bias). Each wave owns one
// 16-wide N-tile (d = wave*16 + lane%16); 4 M-tiles x KDIM/32 K-chunks. ----
template<int KDIM>
__device__ __forceinline__ void dense_relu_layer(
    const _Float16* __restrict__ Xin, int in_stride,
    const _Float16* __restrict__ Wh, const float* __restrict__ Bias,
    _Float16* __restrict__ Xout, int out_stride,
    int wave, int lane) {
  const int d = wave * 16 + (lane & 15);
  const float bias = Bias[d];
  v8f acc[4];
#pragma unroll
  for (int m = 0; m < 4; ++m)
#pragma unroll
    for (int r = 0; r < 8; ++r) acc[m][r] = bias;

  constexpr int KC = KDIM / 32;
#pragma unroll
  for (int kc = 0; kc < KC; ++kc) {
    v16h bf = load_b_frag(Wh, KDIM, d, kc, lane);
#pragma unroll
    for (int m = 0; m < 4; ++m) {
      v16h af = load_a_frag(Xin, in_stride, m * 16 + (lane & 15), kc * 32, lane);
      acc[m] = __builtin_amdgcn_wmma_f32_16x16x32_f16(
          false, af, false, bf, (short)0, acc[m], false, false);
    }
  }
#pragma unroll
  for (int m = 0; m < 4; ++m)
#pragma unroll
    for (int r = 0; r < 8; ++r) {
      int row = m * 16 + r + ((lane >> 4) << 3);
      float v = acc[m][r];
      v = v > 0.f ? v : 0.f;
      Xout[row * out_stride + d] = (_Float16)v;
    }
}

// ---- Final attention layer fused with att3 dot: logits[row] += att3_w[d] *
// relu(att2 @ a + b)[row][d] via ds_add_f32 atomics (no a2-out materialization). ----
template<int KDIM>
__device__ __forceinline__ void dense_relu_dot_logits(
    const _Float16* __restrict__ Xin, int in_stride,
    const _Float16* __restrict__ Wh, const float* __restrict__ Bias,
    const float* __restrict__ a3w, float* logits,
    int wave, int lane) {
  const int d = wave * 16 + (lane & 15);
  const float bias = Bias[d];
  const float wd = a3w[d];
  v8f acc[4];
#pragma unroll
  for (int m = 0; m < 4; ++m)
#pragma unroll
    for (int r = 0; r < 8; ++r) acc[m][r] = bias;

  constexpr int KC = KDIM / 32;
#pragma unroll
  for (int kc = 0; kc < KC; ++kc) {
    v16h bf = load_b_frag(Wh, KDIM, d, kc, lane);
#pragma unroll
    for (int m = 0; m < 4; ++m) {
      v16h af = load_a_frag(Xin, in_stride, m * 16 + (lane & 15), kc * 32, lane);
      acc[m] = __builtin_amdgcn_wmma_f32_16x16x32_f16(
          false, af, false, bf, (short)0, acc[m], false, false);
    }
  }
#pragma unroll
  for (int m = 0; m < 4; ++m)
#pragma unroll
    for (int r = 0; r < 8; ++r) {
      int row = m * 16 + r + ((lane >> 4) << 3);
      float v = acc[m][r];
      v = v > 0.f ? v : 0.f;
      atomicAdd(&logits[row], wd * v);
    }
}

__global__ __launch_bounds__(256)
void vc_aggregator_kernel(
    const int* __restrict__ nodes, const int* __restrict__ hvc,
    const int* __restrict__ hr,
    const float* __restrict__ c2e, const float* __restrict__ v2e,
    const float* __restrict__ r2e,
    const _Float16* __restrict__ wsh,             // converted f16 weights
    const float* __restrict__ w1b, const float* __restrict__ w2b,
    const float* __restrict__ a1b, const float* __restrict__ a2b,
    const float* __restrict__ a3w, const float* __restrict__ a3b,
    float* __restrict__ out) {
  __shared__ _Float16 X0[MROWS * X0_STRIDE];   // 33792 B
  __shared__ _Float16 X1[MROWS * X1_STRIDE];   // 17408 B
  __shared__ _Float16 ubf[DDIM];
  __shared__ int hvc_s[MROWS];
  __shared__ int hr_s[MROWS];
  __shared__ float logits[MROWS];
  __shared__ float wts[MROWS];

  const int b    = blockIdx.x;
  const int tid  = threadIdx.x;
  const int lane = tid & 31;
  const int wave = tid >> 5;

  // ---- Step 0: stage history indices + node rep ----
  if (tid < MROWS) {
    hvc_s[tid] = (tid < LDIM) ? hvc[b * LDIM + tid] : 0;
    hr_s[tid]  = (tid < LDIM) ? hr[b * LDIM + tid] : 0;
  }
  const int node = nodes[b];
  if (tid < DDIM) ubf[tid] = (_Float16)v2e[(size_t)node * DDIM + tid];
  __syncthreads();

  // ---- Step 1: gather embeddings -> X0 = cat(e_vc, e_r), f16, float4 loads ----
  for (int idx = tid; idx < MROWS * 64; idx += 256) {
    int l  = idx >> 6;
    int c  = (idx & 63) << 2;          // column (halves), step 4
    v4h h = (v4h)(_Float16)0.f;
    if (l < LDIM) {
      v4f v;
      if (c < DDIM) v = *(const v4f*)(c2e + (size_t)hvc_s[l] * DDIM + c);
      else          v = *(const v4f*)(r2e + (size_t)hr_s[l] * DDIM + (c - DDIM));
      h[0] = (_Float16)v[0]; h[1] = (_Float16)v[1];
      h[2] = (_Float16)v[2]; h[3] = (_Float16)v[3];
    }
    *(v4h*)(&X0[l * X0_STRIDE + c]) = h;
  }
  __syncthreads();

  // ---- Step 2: x = relu(W_r1 @ X0 + b1)   (K=256) -> X1 ----
  dense_relu_layer<256>(X0, X0_STRIDE, wsh + W1H_OFF, w1b, X1, X1_STRIDE, wave, lane);
  __syncthreads();

  // ---- Step 3: H = relu(W_r2 @ X1 + b2)   (K=128) -> X0[:, :128];
  //              X0[:, 128:256] = u (broadcast node rep) ----
  dense_relu_layer<128>(X1, X1_STRIDE, wsh + W2H_OFF, w2b, X0, X0_STRIDE, wave, lane);
  for (int idx = tid; idx < MROWS * (DDIM / 4); idx += 256) {
    int l = idx >> 5;
    int c = (idx & 31) << 2;
    *(v4h*)(&X0[l * X0_STRIDE + DDIM + c]) = *(const v4h*)(&ubf[c]);
  }
  __syncthreads();

  // ---- Step 4: a = relu(att1 @ X0 + b)    (K=256) -> X1 ----
  dense_relu_layer<256>(X0, X0_STRIDE, wsh + A1H_OFF, a1b, X1, X1_STRIDE, wave, lane);
  if (tid < MROWS) logits[tid] = a3b[0];
  __syncthreads();

  // ---- Step 5: logits += att3_w . relu(att2 @ X1 + b)  (K=128, fused) ----
  dense_relu_dot_logits<128>(X1, X1_STRIDE, wsh + A2H_OFF, a2b, a3w, logits, wave, lane);
  __syncthreads();

  // ---- Step 6: softmax over L=50 (serial, trivially cheap) ----
  if (tid == 0) {
    float mx = -1e30f;
    for (int l = 0; l < LDIM; ++l) mx = fmaxf(mx, logits[l]);
    float s = 0.f;
    for (int l = 0; l < LDIM; ++l) { float e = __expf(logits[l] - mx); wts[l] = e; s += e; }
    float inv = 1.f / s;
    for (int l = 0; l < LDIM; ++l) wts[l] *= inv;
  }
  __syncthreads();

  // ---- Step 7: out[b][d] = sum_l wts[l] * H[l][d]  (H lives in X0[:, :128]) ----
  if (tid < DDIM) {
    float s = 0.f;
    for (int l = 0; l < LDIM; ++l) s += wts[l] * (float)X0[l * X0_STRIDE + tid];
    out[(size_t)b * DDIM + tid] = s;
  }
}

extern "C" void kernel_launch(void* const* d_in, const int* in_sizes, int n_in,
                              void* d_out, int out_size, void* d_ws, size_t ws_size,
                              hipStream_t stream) {
  const int*   nodes = (const int*)d_in[0];
  const int*   hvc   = (const int*)d_in[1];
  const int*   hr    = (const int*)d_in[2];
  const float* c2e   = (const float*)d_in[3];
  const float* v2e   = (const float*)d_in[4];
  const float* r2e   = (const float*)d_in[5];
  const float* w1w   = (const float*)d_in[6];
  const float* w1b   = (const float*)d_in[7];
  const float* w2w   = (const float*)d_in[8];
  const float* w2b   = (const float*)d_in[9];
  const float* a1w   = (const float*)d_in[10];
  const float* a1b   = (const float*)d_in[11];
  const float* a2w   = (const float*)d_in[12];
  const float* a2b   = (const float*)d_in[13];
  const float* a3w   = (const float*)d_in[14];
  const float* a3b   = (const float*)d_in[15];
  float* out = (float*)d_out;

  _Float16* wsh = (_Float16*)d_ws;   // WSH_TOTAL halves = 192KB of scratch

  // Convert weights f32 -> f16 once per launch (stream-ordered, capture-safe).
  {
    const int n1 = 128 * 256 / 4, n2 = 128 * 128 / 4;
    cvt_weights_kernel<<<(n1 + 255) / 256, 256, 0, stream>>>(w1w, wsh + W1H_OFF, n1);
    cvt_weights_kernel<<<(n2 + 255) / 256, 256, 0, stream>>>(w2w, wsh + W2H_OFF, n2);
    cvt_weights_kernel<<<(n1 + 255) / 256, 256, 0, stream>>>(a1w, wsh + A1H_OFF, n1);
    cvt_weights_kernel<<<(n2 + 255) / 256, 256, 0, stream>>>(a2w, wsh + A2H_OFF, n2);
  }

  vc_aggregator_kernel<<<BDIM, 256, 0, stream>>>(
      nodes, hvc, hr, c2e, v2e, r2e, wsh,
      w1b, w2b, a1b, a2b, a3w, a3b, out);
}
